// SNN_54348516164146
// MI455X (gfx1250) — matrix operand order repaired and checked by reading, not compiled, not run on hardware
//
#include <hip/hip_runtime.h>

// ---------------------------------------------------------------------------
// EP free-phase relaxation on gfx1250: bf16 WMMA + Tensor Data Mover staging.
// State s is always in [0,1] after the clipped update, so rho(s) == s; we keep
// a bf16 mirror of each state in global memory and TDM-load it into LDS.
// ---------------------------------------------------------------------------

typedef __bf16 bf16;
typedef __attribute__((ext_vector_type(8)))  bf16  v8bf;
typedef __attribute__((ext_vector_type(16))) bf16  v16bf;
typedef __attribute__((ext_vector_type(8)))  float v8f;
typedef __attribute__((ext_vector_type(4)))  unsigned int u32x4;
typedef __attribute__((ext_vector_type(8)))  int i32x8;
typedef __attribute__((ext_vector_type(4)))  int i32x4;

#define DT 0.2f

__device__ __forceinline__ float clamp01(float x) { return fminf(fmaxf(x, 0.f), 1.f); }

__device__ __forceinline__ v16bf catbf(v8bf lo, v8bf hi) {
  return __builtin_shufflevector(lo, hi, 0,1,2,3,4,5,6,7,8,9,10,11,12,13,14,15);
}

__device__ __forceinline__ v8f wmma_bf16(v16bf a, v16bf b, v8f c) {
  return __builtin_amdgcn_wmma_f32_16x16x32_bf16(false, a, false, b, (short)0, c, false, false);
}

// ---------------------------------------------------------------------------
// TDM 2D tile load: global (bf16, row stride stride0 elements) -> LDS.
// D# per CDNA5 ISA: group0 = {flags, lds_addr,
// global_addr, type=2}; group1 packs data_size=2B, tensor/tile dims, stride.
// Groups 2/3 zero (2D tensor). Tracked by TENSORcnt.
// amdgpu-toolchain (clang-23): 6-arg builtin (g0, g1, g2, g3, g4, cpol).
// ---------------------------------------------------------------------------
__device__ __forceinline__ void tdm_load_2d(unsigned lds_addr, const void* gptr,
                                            unsigned tensor_d0, unsigned tensor_d1,
                                            unsigned stride0,
                                            unsigned tile_d0, unsigned tile_d1) {
  unsigned long long ga = (unsigned long long)gptr;
  u32x4 g0;
  g0[0] = 1u;                                                  // count=1
  g0[1] = lds_addr;                                            // LDS byte addr
  g0[2] = (unsigned)(ga & 0xffffffffu);                        // global addr lo
  g0[3] = (unsigned)((ga >> 32) & 0x01ffffffu) | (2u << 30);   // addr hi | type=2
  i32x8 g1;
  g1[0] = (int)(1u << 16);                                     // data_size=1 -> 2B
  g1[1] = (int)((tensor_d0 & 0xffffu) << 16);                  // dim0[15:0] @63:48
  g1[2] = (int)(((tensor_d0 >> 16) & 0xffffu) |                // dim0[31:16]
                ((tensor_d1 & 0xffffu) << 16));                // dim1[15:0]
  g1[3] = (int)(((tensor_d1 >> 16) & 0xffffu) |                // dim1[31:16]
                ((tile_d0 & 0xffffu) << 16));                  // tile_dim0 @127:112
  g1[4] = (int)(tile_d1 & 0xffffu);                            // tile_dim1 (dim2=0)
  g1[5] = (int)stride0;                                        // dim0 stride lo32
  g1[6] = 0;
  g1[7] = 0;
  i32x4 gz4 = {0, 0, 0, 0};
  i32x8 gz8 = {0, 0, 0, 0, 0, 0, 0, 0};
  __builtin_amdgcn_tensor_load_to_lds(g0, g1, gz4, gz4, gz8, 0);
}

// ---------------------------------------------------------------------------
// Pack weights into WMMA B-fragment order (bf16, 32(K) x 16(N) tiles).
//   element e of lane l in tile (nt,kt): k = kt*32+(l>>4)*8+e+(e>=8?8:0), n = nt*16+(l&15)
//   value = transpose ? W[k*ldw+n] : W[n*ldw+k]  (0 outside Kv/Nv)
// tile = nt*nKt + kt; 512 bf16/tile, 16 contiguous per lane.
// ---------------------------------------------------------------------------
__global__ __launch_bounds__(256) void pack_B(const float* __restrict__ W, bf16* __restrict__ P,
                                              int transpose, int nKt, int nNt, int ldw,
                                              int Kv, int Nv) {
  int t = blockIdx.x * 256 + threadIdx.x;
  int total = nKt * nNt * 32;
  if (t >= total) return;
  int lane = t & 31;
  int tile = t >> 5;
  int nt = tile / nKt;
  int kt = tile % nKt;
  int kbase = kt * 32 + ((lane >> 4) << 3);
  int n = nt * 16 + (lane & 15);
  __align__(32) bf16 vals[16];
#pragma unroll
  for (int e = 0; e < 16; ++e) {
    int k = kbase + e + ((e >> 3) << 3);
    float f = 0.f;
    if (k < Kv && n < Nv)
      f = transpose ? W[(size_t)k * ldw + n] : W[(size_t)n * ldw + k];
    vals[e] = (bf16)f;
  }
  *(v16bf*)(P + (size_t)tile * 512 + lane * 16) = *(const v16bf*)vals;
}

// ---------------------------------------------------------------------------
// Initialize f32 state buffers (s0 padded [256][16]) and bf16 activation
// mirrors (r0 padded [256][32], cols >=16 stay zero forever).
// ---------------------------------------------------------------------------
__global__ __launch_bounds__(256) void init_states(const float* __restrict__ s0,
                                                   const float* __restrict__ s1,
                                                   const float* __restrict__ s2,
                                                   float* __restrict__ s0b,
                                                   float* __restrict__ s1b,
                                                   float* __restrict__ s2b,
                                                   bf16* __restrict__ r0g,
                                                   bf16* __restrict__ r1g,
                                                   bf16* __restrict__ r2g) {
  int t = blockIdx.x * 256 + threadIdx.x;
  if (t < 4096) { int r = t >> 4, c = t & 15; s0b[t] = (c < 10) ? s0[r * 10 + c] : 0.f; return; }
  t -= 4096;
  if (t < 262144) { s1b[t] = s1[t]; return; }
  t -= 262144;
  if (t < 262144) { s2b[t] = s2[t]; return; }
  t -= 262144;
  if (t < 8192) {
    int r = t >> 5, c = t & 31;
    float v = (c < 10) ? clamp01(s0[r * 10 + c]) : 0.f;
    r0g[t] = (bf16)v;
    return;
  }
  t -= 8192;
  if (t < 262144) { r1g[t] = (bf16)clamp01(s1[t]); return; }
  t -= 262144;
  if (t < 262144) { r2g[t] = (bf16)clamp01(s2[t]); }
}

// ---------------------------------------------------------------------------
// inp2 = rho(data) @ W4.T + b4 : [256,784]x[784,1024] -> [256,1024], K pad 800.
// Runs once; manual f32->bf16 staging (needs the conversion anyway).
// ---------------------------------------------------------------------------
__global__ __launch_bounds__(128) void inp2_gemm(const float* __restrict__ data,
                                                 const bf16* __restrict__ P4,
                                                 const float* __restrict__ b4,
                                                 float* __restrict__ inp2) {
  __shared__ __align__(16) bf16 ldsA[16 * 800];
  int tid = threadIdx.x, lane = tid & 31, wave = tid >> 5;
  int rb = blockIdx.x >> 4, cg = blockIdx.x & 15;

  for (int i = tid; i < 16 * 200; i += 128) {
    int r = i / 200, c4 = i % 200;
    float4 f = make_float4(0.f, 0.f, 0.f, 0.f);
    if (c4 < 196) f = *(const float4*)(data + (size_t)(rb * 16 + r) * 784 + c4 * 4);
    bf16* d = &ldsA[r * 800 + c4 * 4];
    d[0] = (bf16)clamp01(f.x); d[1] = (bf16)clamp01(f.y);
    d[2] = (bf16)clamp01(f.z); d[3] = (bf16)clamp01(f.w);
  }
  __syncthreads();

  int m = lane & 15, koff = (lane >> 4) << 3, n = lane & 15;
  int ntile = cg * 4 + wave, colbase = ntile * 16;
  const bf16* Pt = P4 + (size_t)ntile * 25 * 512;
  v8f acc = {};
  for (int kt = 0; kt < 25; ++kt) {
    int kb = kt * 32;
    v8bf lo = *(const v8bf*)&ldsA[m * 800 + kb + koff];
    v8bf hi = *(const v8bf*)&ldsA[m * 800 + kb + koff + 16];
    v16bf b = *(const v16bf*)(Pt + (size_t)kt * 512 + lane * 16);
    acc = wmma_bf16(catbf(lo, hi), b, acc);
  }
#pragma unroll
  for (int v = 0; v < 8; ++v) {
    int row = rb * 16 + ((lane >> 4) << 3) + v;
    int col = colbase + n;
    inp2[(size_t)row * 1024 + col] = acc[v] + b4[col];
  }
}

// ---------------------------------------------------------------------------
// One fused Euler step. 136 blocks x 128 threads (4 wave32s).
//   blocks   0.. 63 : layer1 (r2@W2.T + r0@W1.T + b2)    32 rows x 128 cols
//   blocks  64..127 : layer2 (r1@W3.T + inp2)            32 rows x 128 cols
//   blocks 128..135 : layer0 (r1@W0.T + b0, split-K)     32 rows x 16 cols
// Each wave owns a 32x32 register tile (2x2 WMMA accumulators).
// Activations staged into LDS by the Tensor Data Mover (wave 0 issues).
// ---------------------------------------------------------------------------
__global__ __launch_bounds__(128) void ep_step(
    const float* __restrict__ s0_in, const float* __restrict__ s1_in, const float* __restrict__ s2_in,
    const bf16* __restrict__ r0_in, const bf16* __restrict__ r1_in, const bf16* __restrict__ r2_in,
    float* __restrict__ s0_out, float* __restrict__ s1_out, float* __restrict__ s2_out,
    bf16* __restrict__ r0_out, bf16* __restrict__ r1_out, bf16* __restrict__ r2_out,
    const bf16* __restrict__ P1, const bf16* __restrict__ P1x,
    const bf16* __restrict__ P0, const bf16* __restrict__ P2,
    const float* __restrict__ inp2, const float* __restrict__ b0v, const float* __restrict__ b2v) {
  __shared__ __align__(16) unsigned char smem[32 * 1024 * 2 + 2048];
  bf16* ldsA   = (bf16*)smem;               // 32 rows x 1024 cols (64 KB)
  bf16* ldsR0  = (bf16*)(smem + 65536);     // 32 rows x 32 cols  (2 KB)
  float* ldsRed = (float*)smem;             // layer0 reduction, aliases ldsA

  int bid = blockIdx.x, tid = threadIdx.x, lane = tid & 31, wave = tid >> 5;
  int layer, rb, cg;
  if (bid < 64)       { layer = 1; rb = bid >> 3;        cg = bid & 7; }
  else if (bid < 128) { layer = 2; rb = (bid - 64) >> 3; cg = (bid - 64) & 7; }
  else                { layer = 0; rb = bid - 128;       cg = 0; }

  const bf16* act = (layer == 1) ? r2_in : r1_in;
  if (tid == 0) {
    tdm_load_2d((unsigned)(size_t)ldsA, act + (size_t)rb * 32 * 1024,
                1024, 256, 1024, 1024, 32);
    if (layer == 1)
      tdm_load_2d((unsigned)(size_t)ldsR0, r0_in + (size_t)rb * 32 * 32,
                  32, 256, 32, 32, 32);
    __builtin_amdgcn_s_wait_tensorcnt(0);
  }
  __syncthreads();

  int m = lane & 15, koff = (lane >> 4) << 3, n = lane & 15;

  if (layer != 0) {
    int nt0 = cg * 8 + wave * 2;  // two adjacent 16-col tiles: nt0, nt0+1
    const bf16* Pm  = (layer == 1) ? P1 : P2;
    const bf16* Pt0 = Pm + (size_t)nt0 * 32 * 512;
    const bf16* Pt1 = Pm + (size_t)(nt0 + 1) * 32 * 512;
    v8f acc00 = {}, acc01 = {}, acc10 = {}, acc11 = {};
#pragma unroll 2
    for (int kt = 0; kt < 32; ++kt) {
      int kb = kt * 32;
      v8bf a0l = *(const v8bf*)&ldsA[(size_t)m * 1024 + kb + koff];
      v8bf a0h = *(const v8bf*)&ldsA[(size_t)m * 1024 + kb + koff + 16];
      v8bf a1l = *(const v8bf*)&ldsA[(size_t)(m + 16) * 1024 + kb + koff];
      v8bf a1h = *(const v8bf*)&ldsA[(size_t)(m + 16) * 1024 + kb + koff + 16];
      v16bf a0 = catbf(a0l, a0h), a1 = catbf(a1l, a1h);
      v16bf b0 = *(const v16bf*)(Pt0 + (size_t)kt * 512 + lane * 16);
      v16bf b1 = *(const v16bf*)(Pt1 + (size_t)kt * 512 + lane * 16);
      acc00 = wmma_bf16(a0, b0, acc00);
      acc01 = wmma_bf16(a0, b1, acc01);
      acc10 = wmma_bf16(a1, b0, acc10);
      acc11 = wmma_bf16(a1, b1, acc11);
    }
    if (layer == 1) {  // + r0 @ W1.T (K=10 padded to 32)
      v8bf a0l = *(const v8bf*)&ldsR0[m * 32 + koff];
      v8bf a0h = *(const v8bf*)&ldsR0[m * 32 + koff + 16];
      v8bf a1l = *(const v8bf*)&ldsR0[(m + 16) * 32 + koff];
      v8bf a1h = *(const v8bf*)&ldsR0[(m + 16) * 32 + koff + 16];
      v16bf a0 = catbf(a0l, a0h), a1 = catbf(a1l, a1h);
      v16bf b0 = *(const v16bf*)(P1x + (size_t)nt0 * 512 + lane * 16);
      v16bf b1 = *(const v16bf*)(P1x + (size_t)(nt0 + 1) * 512 + lane * 16);
      acc00 = wmma_bf16(a0, b0, acc00);
      acc01 = wmma_bf16(a0, b1, acc01);
      acc10 = wmma_bf16(a1, b0, acc10);
      acc11 = wmma_bf16(a1, b1, acc11);
    }
    const float* sin = (layer == 1) ? s1_in : s2_in;
    float* sout      = (layer == 1) ? s1_out : s2_out;
    bf16* rout       = (layer == 1) ? r1_out : r2_out;
    v8f accA[2][2] = { { acc00, acc01 }, { acc10, acc11 } };
#pragma unroll
    for (int mt = 0; mt < 2; ++mt)
#pragma unroll
      for (int nn = 0; nn < 2; ++nn)
#pragma unroll
        for (int v = 0; v < 8; ++v) {
          int row = rb * 32 + mt * 16 + ((lane >> 4) << 3) + v;
          int col = (nt0 + nn) * 16 + n;
          float extra = (layer == 1) ? b2v[col] : inp2[(size_t)row * 1024 + col];
          float s = sin[(size_t)row * 1024 + col];
          float ns = clamp01(s + DT * (accA[mt][nn][v] + extra - s));
          sout[(size_t)row * 1024 + col] = ns;
          rout[(size_t)row * 1024 + col] = (bf16)ns;
        }
  } else {
    // Layer 0: single N-tile (cols >=10 padded), split-K across the 4 waves.
    v8f acc0 = {}, acc1 = {};
    for (int kt = wave; kt < 32; kt += 4) {
      int kb = kt * 32;
      v8bf a0l = *(const v8bf*)&ldsA[(size_t)m * 1024 + kb + koff];
      v8bf a0h = *(const v8bf*)&ldsA[(size_t)m * 1024 + kb + koff + 16];
      v8bf a1l = *(const v8bf*)&ldsA[(size_t)(m + 16) * 1024 + kb + koff];
      v8bf a1h = *(const v8bf*)&ldsA[(size_t)(m + 16) * 1024 + kb + koff + 16];
      v16bf b = *(const v16bf*)(P0 + (size_t)kt * 512 + lane * 16);
      acc0 = wmma_bf16(catbf(a0l, a0h), b, acc0);
      acc1 = wmma_bf16(catbf(a1l, a1h), b, acc1);
    }
    __syncthreads();  // everyone done reading ldsA before aliasing as ldsRed
#pragma unroll
    for (int v = 0; v < 8; ++v) {
      ldsRed[((wave * 2 + 0) * 32 + lane) * 8 + v] = acc0[v];
      ldsRed[((wave * 2 + 1) * 32 + lane) * 8 + v] = acc1[v];
    }
    __syncthreads();
    if (wave == 0) {
#pragma unroll
      for (int mt = 0; mt < 2; ++mt)
#pragma unroll
        for (int v = 0; v < 8; ++v) {
          float sum = 0.f;
#pragma unroll
          for (int w = 0; w < 4; ++w) sum += ldsRed[((w * 2 + mt) * 32 + lane) * 8 + v];
          int row = rb * 32 + mt * 16 + ((lane >> 4) << 3) + v;
          int col = n;
          float bias = (col < 10) ? b0v[col] : 0.f;
          float s = s0_in[row * 16 + col];
          float ns = clamp01(s + DT * (sum + bias - s));
          s0_out[row * 16 + col] = ns;
          r0_out[row * 32 + col] = (bf16)ns;
        }
    }
  }
}

// ---------------------------------------------------------------------------
// Emit (s0[256,10], s1[256,1024], s2[256,1024]) concatenated into d_out.
// ---------------------------------------------------------------------------
__global__ __launch_bounds__(256) void final_copy(const float* __restrict__ s0b,
                                                  const float* __restrict__ s1b,
                                                  const float* __restrict__ s2b,
                                                  float* __restrict__ out) {
  int t = blockIdx.x * 256 + threadIdx.x;
  if (t < 2560) { int r = t / 10, c = t % 10; out[t] = s0b[r * 16 + c]; return; }
  t -= 2560;
  if (t < 262144) { out[2560 + t] = s1b[t]; return; }
  t -= 262144;
  if (t < 262144) { out[2560 + 262144 + t] = s2b[t]; }
}

extern "C" void kernel_launch(void* const* d_in, const int* in_sizes, int n_in,
                              void* d_out, int out_size, void* d_ws, size_t ws_size,
                              hipStream_t stream) {
  (void)in_sizes; (void)n_in; (void)out_size; (void)ws_size;
  const float* data = (const float*)d_in[0];
  const float* s0i  = (const float*)d_in[1];
  const float* s1i  = (const float*)d_in[2];
  const float* s2i  = (const float*)d_in[3];
  const float* W0   = (const float*)d_in[4];
  const float* b0   = (const float*)d_in[5];
  const float* W2   = (const float*)d_in[7];
  const float* b2   = (const float*)d_in[8];
  const float* W4   = (const float*)d_in[10];
  const float* b4   = (const float*)d_in[11];

  char* ws = (char*)d_ws;
  size_t off = 0;
  auto alloc = [&](size_t bytes) -> void* {
    void* p = ws + off;
    off += (bytes + 255) & ~(size_t)255;
    return p;
  };
  float* s0b[2] = { (float*)alloc(4096 * 4),   (float*)alloc(4096 * 4) };
  float* s1b[2] = { (float*)alloc(262144 * 4), (float*)alloc(262144 * 4) };
  float* s2b[2] = { (float*)alloc(262144 * 4), (float*)alloc(262144 * 4) };
  bf16* r0b[2]  = { (bf16*)alloc(8192 * 2),    (bf16*)alloc(8192 * 2) };
  bf16* r1b[2]  = { (bf16*)alloc(262144 * 2),  (bf16*)alloc(262144 * 2) };
  bf16* r2b[2]  = { (bf16*)alloc(262144 * 2),  (bf16*)alloc(262144 * 2) };
  float* inp2   = (float*)alloc(262144 * 4);
  bf16* P1  = (bf16*)alloc((size_t)2048 * 512 * 2);  // W2, B(k,n)=W2[n,k]
  bf16* P2  = (bf16*)alloc((size_t)2048 * 512 * 2);  // W2, B(k,n)=W2[k,n]
  bf16* P0  = (bf16*)alloc((size_t)32 * 512 * 2);    // W0, B(k,n)=W0[n,k] (N pad 16)
  bf16* P1x = (bf16*)alloc((size_t)64 * 512 * 2);    // W0, B(k,n)=W0[k,n] (K pad 32)
  bf16* P4  = (bf16*)alloc((size_t)1600 * 512 * 2);  // W4, B(k,n)=W4[n,k] (K pad 800)

  pack_B<<<(2048 * 32 + 255) / 256, 256, 0, stream>>>(W2, P1, 0, 32, 64, 1024, 1024, 1024);
  pack_B<<<(2048 * 32 + 255) / 256, 256, 0, stream>>>(W2, P2, 1, 32, 64, 1024, 1024, 1024);
  pack_B<<<(32 * 32 + 255) / 256,   256, 0, stream>>>(W0, P0, 0, 32, 1, 1024, 1024, 10);
  pack_B<<<(64 * 32 + 255) / 256,   256, 0, stream>>>(W0, P1x, 1, 1, 64, 1024, 10, 1024);
  pack_B<<<(1600 * 32 + 255) / 256, 256, 0, stream>>>(W4, P4, 0, 25, 64, 784, 784, 1024);

  init_states<<<(1060864 + 255) / 256, 256, 0, stream>>>(s0i, s1i, s2i,
                                                         s0b[0], s1b[0], s2b[0],
                                                         r0b[0], r1b[0], r2b[0]);
  inp2_gemm<<<256, 128, 0, stream>>>(data, P4, b4, inp2);

  for (int t = 0; t < 20; ++t) {
    int a = t & 1, b = a ^ 1;
    ep_step<<<136, 128, 0, stream>>>(s0b[a], s1b[a], s2b[a],
                                     r0b[a], r1b[a], r2b[a],
                                     s0b[b], s1b[b], s2b[b],
                                     r0b[b], r1b[b], r2b[b],
                                     P1, P1x, P0, P2, inp2, b0, b2);
  }
  // 20 steps: final state lands in buffer index 0.
  final_copy<<<(526848 + 255) / 256, 256, 0, stream>>>(s0b[0], s1b[0], s2b[0], (float*)d_out);
}
